// GN_NN_16844861735301
// MI455X (gfx1250) — matrix operand order repaired and checked by reading, not compiled, not run on hardware
//
#include <hip/hip_runtime.h>

typedef __attribute__((ext_vector_type(16))) _Float16 v16h;
typedef __attribute__((ext_vector_type(8)))  _Float16 v8h;
typedef __attribute__((ext_vector_type(8)))  float    v8f;
typedef __attribute__((ext_vector_type(4)))  int      i32x4;

#define LDS_FENCE() asm volatile("s_wait_dscnt 0x0" ::: "memory")

// ---------------------------------------------------------------------------
// Async global->LDS 16B copy (gfx1250 ASYNC LDS DMA, tracked by ASYNCcnt).
// Builtin prototype (from compiler diagnostic): (i32x4 __global*, i32x4 __local*, imm, imm).
// Guarded: falls back to a plain register copy if the builtin is absent.
// ---------------------------------------------------------------------------
__device__ __forceinline__ void async_copy16(const void* g, void* l) {
#if __has_builtin(__builtin_amdgcn_global_load_async_to_lds_b128)
  __builtin_amdgcn_global_load_async_to_lds_b128(
      (__attribute__((address_space(1))) i32x4*)(void*)g,
      (__attribute__((address_space(3))) i32x4*)l, 0, 0);
#else
  *(v8h*)l = *(const v8h*)g;
#endif
}

__device__ __forceinline__ void async_join() {
#if __has_builtin(__builtin_amdgcn_s_wait_asynccnt)
  __builtin_amdgcn_s_wait_asynccnt(0);
#endif
}

// ---------------------------------------------------------------------------
// Fragment helpers (wave32, 16x16x32 f16 WMMA)
// A layout (16-bit 16x32): lane = half*16+m holds row m; VGPR0-3 = K half*8+0..7,
// VGPR4-7 = K 16+half*8+0..7  -> two contiguous 16B chunks per lane.
// B fragments pre-swizzled: fragment (kt,nt), lane l, element j =
//   W[kt*32 + (j<8 ? (l>>4)*8+j : 16+(l>>4)*8+j-8)][nt*16 + (l&15)], contiguous 32B per lane.
// C/D layout: lane l, reg r -> row m = r + 8*(l>>4), col n = (l&15).
// ---------------------------------------------------------------------------
__device__ __forceinline__ v16h cat8(v8h lo, v8h hi) {
  return __builtin_shufflevector(lo, hi, 0,1,2,3,4,5,6,7,8,9,10,11,12,13,14,15);
}

template<int KT, int NT>
__device__ __forceinline__ void wmma_gemm(const _Float16* A, const _Float16* Wswz,
                                          v8f (&acc)[NT], int lane)
{
  const int D    = KT * 32;
  const int half = (lane >> 4) & 1;
  const int m    = lane & 15;
#pragma unroll
  for (int kt = 0; kt < KT; ++kt) {
    const _Float16* rowp = A + m * D + kt * 32;
    v8h lo = *reinterpret_cast<const v8h*>(rowp + half * 8);
    v8h hi = *reinterpret_cast<const v8h*>(rowp + 16 + half * 8);
    v16h a = cat8(lo, hi);
    const v16h* Wf = reinterpret_cast<const v16h*>(Wswz) + (size_t)(kt * NT) * 32;
#pragma unroll
    for (int nt = 0; nt < NT; ++nt) {
      v16h b = Wf[nt * 32 + lane];
      acc[nt] = __builtin_amdgcn_wmma_f32_16x16x32_f16(
          false, a, false, b, (short)0, acc[nt], false, false);
    }
  }
}

template<int NT>
__device__ __forceinline__ void bias_act(v8f (&acc)[NT], const float* bias,
                                         int lane, bool relu)
{
  const int n = lane & 15;
#pragma unroll
  for (int nt = 0; nt < NT; ++nt) {
    float bn = bias[nt * 16 + n];
#pragma unroll
    for (int r = 0; r < 8; ++r) {
      float v = acc[nt][r] + bn;
      if (relu) v = v > 0.f ? v : 0.f;
      acc[nt][r] = v;
    }
  }
}

// LayerNorm across full row (D = NT*16) in registers: bias add, then
// cross-lane (width 16) shuffle reductions for mean / mean-of-squares.
template<int NT>
__device__ __forceinline__ void ln_inplace(v8f (&acc)[NT], const float* bias,
                                           const float* g, const float* bln, int lane)
{
  const int n = lane & 15;
  const float invD = 1.0f / (float)(NT * 16);
#pragma unroll
  for (int nt = 0; nt < NT; ++nt) {
    float bn = bias[nt * 16 + n];
#pragma unroll
    for (int r = 0; r < 8; ++r) acc[nt][r] += bn;
  }
#pragma unroll
  for (int r = 0; r < 8; ++r) {
    float s = 0.f, ss = 0.f;
#pragma unroll
    for (int nt = 0; nt < NT; ++nt) { float v = acc[nt][r]; s += v; ss += v * v; }
#pragma unroll
    for (int mask = 1; mask < 16; mask <<= 1) {
      s  += __shfl_xor(s,  mask, 16);
      ss += __shfl_xor(ss, mask, 16);
    }
    float mu   = s * invD;
    float var  = ss * invD - mu * mu;
    float rstd = rsqrtf(var + 1e-5f);
#pragma unroll
    for (int nt = 0; nt < NT; ++nt) {
      int nn = nt * 16 + n;
      acc[nt][r] = (acc[nt][r] - mu) * rstd * g[nn] + bln[nn];
    }
  }
}

template<int NT>
__device__ __forceinline__ void store_f16(const v8f (&acc)[NT], _Float16* S,
                                          int Dout, int colOff, int lane)
{
  const int half = (lane >> 4) & 1, n = lane & 15;
#pragma unroll
  for (int nt = 0; nt < NT; ++nt)
#pragma unroll
    for (int r = 0; r < 8; ++r)
      S[(r + 8 * half) * Dout + colOff + nt * 16 + n] = (_Float16)acc[nt][r];
}

// ---------------------------------------------------------------------------
// Workspace layout (f16 element offsets):
//   edge weights (contiguous 24576 el): mm_w2 @0, ms_w1 @8192, ms_w2 @16384
//   node weights (contiguous 98304 el): enc_w1 @24576, enc_w2 @32768,
//     encs_w1 @49152, encs_w2 @65536, up_w1 @81920, up_w2 @106496
//   byte 262144: aggr [nNodes x 64] f32, then deg [nNodes] f32
// ---------------------------------------------------------------------------
#define OFF_MM_W2   0
#define OFF_MS_W1   8192
#define OFF_MS_W2   16384
#define OFF_ENC_W1  24576
#define OFF_ENC_W2  32768
#define OFF_ENCS_W1 49152
#define OFF_ENCS_W2 65536
#define OFF_UP_W1   81920
#define OFF_UP_W2   106496
#define AGGR_BYTE_OFF 262144

__global__ void zero_kernel(float* p, long long n)
{
  long long i = (long long)blockIdx.x * blockDim.x + threadIdx.x;
  long long stride = (long long)gridDim.x * blockDim.x;
  for (; i < n; i += stride) p[i] = 0.0f;
}

// f32 [K,N] row-major -> f16 B-fragment-swizzled
__global__ void swizzle_kernel(const float* __restrict__ src, _Float16* __restrict__ dst,
                               int KT, int NT, int N)
{
  int tid  = blockIdx.x * blockDim.x + threadIdx.x;
  int lane = tid & 31;
  int f    = tid >> 5;
  if (f >= KT * NT) return;
  int kt = f / NT, nt = f % NT;
  int half = (lane >> 4) & 1, n = lane & 15;
#pragma unroll
  for (int j = 0; j < 16; ++j) {
    int k = kt * 32 + (j < 8 ? half * 8 + j : 16 + half * 8 + (j - 8));
    dst[((size_t)f * 32 + lane) * 16 + j] = (_Float16)src[(size_t)k * N + nt * 16 + n];
  }
}

// ---------------------------------------------------------------------------
// Edge kernel: persistent waves; weights async-staged to LDS once per block.
// Per 16-edge tile (one wave):
//   h1 = relu(e @ mm_w1 + b1)   (K=3 scalar FMA)
//   m  = LN(h1 @ mm_w2 + b2)    (WMMA 4x4, B from LDS)
//   h2 = relu(m @ ms_w1 + b)    (WMMA 2x8)
//   m2 = h2 @ ms_w2 + b         (WMMA 4x4) -> atomicAdd aggr[dst]; deg[src]+=1
// LDS: weights 49152B, then 4 waves x 6144B act buffers (73728B dynamic).
// ---------------------------------------------------------------------------
#define EW 4
__global__ void __launch_bounds__(128)
edge_kernel(const float* __restrict__ edge_attr, const int* __restrict__ ei,
            const float* __restrict__ mm_w1, const float* __restrict__ mm_b1,
            const float* __restrict__ mm_b2,
            const float* __restrict__ n2g, const float* __restrict__ n2b,
            const float* __restrict__ ms_b1, const float* __restrict__ ms_b2,
            const _Float16* __restrict__ zW,   // ws base: 24576 contiguous f16
            float* __restrict__ aggr, float* __restrict__ deg, int nEdges)
{
  extern __shared__ __align__(16) char smem[];
  _Float16* lds = (_Float16*)smem;
  const int tid = threadIdx.x;

  // Stage all edge-path weights: 49152 B = 3072 x 16B chunks.
  for (int c = tid; c < 3072; c += 128)
    async_copy16((const char*)zW + (size_t)c * 16, smem + (size_t)c * 16);
  async_join();
  __syncthreads();

  const _Float16* w_mm_w2 = lds;
  const _Float16* w_ms_w1 = lds + 8192;
  const _Float16* w_ms_w2 = lds + 16384;

  const int wave = tid >> 5, lane = tid & 31;
  _Float16* b0 = lds + 24576 + wave * 3072;   // 16x128 f16
  _Float16* b1 = b0 + 2048;                   // 16x64  f16
  const int half = (lane >> 4) & 1, n = lane & 15;
  const int nTiles = nEdges >> 4;
  const int* dst = ei + nEdges;

  for (int tile = blockIdx.x * EW + wave; tile < nTiles; tile += gridDim.x * EW) {
    const int e0 = tile * 16;

    if (lane < 16) atomicAdd(&deg[ei[e0 + lane]], 1.0f);

    // layer 1: 16x3 @ 3x128 (+bias, relu) -> b0
    for (int t = lane; t < 16 * 128; t += 32) {
      int m = t >> 7, c = t & 127;
      const float* ea = edge_attr + (size_t)(e0 + m) * 3;
      float v = mm_b1[c] + ea[0] * mm_w1[c] + ea[1] * mm_w1[128 + c] + ea[2] * mm_w1[256 + c];
      b0[t] = (_Float16)(v > 0.f ? v : 0.f);
    }
    LDS_FENCE();

    // layer 2: [16,128]@[128,64] + bias, LayerNorm -> b1
    v8f acc4[4];
#pragma unroll
    for (int i = 0; i < 4; ++i) acc4[i] = 0.0f;
    wmma_gemm<4, 4>(b0, w_mm_w2, acc4, lane);
    ln_inplace<4>(acc4, mm_b2, n2g, n2b, lane);
    store_f16<4>(acc4, b1, 64, 0, lane);
    LDS_FENCE();

    // layer 3: [16,64]@[64,128] + bias, relu -> b0
    v8f acc8[8];
#pragma unroll
    for (int i = 0; i < 8; ++i) acc8[i] = 0.0f;
    wmma_gemm<2, 8>(b1, w_ms_w1, acc8, lane);
    bias_act<8>(acc8, ms_b1, lane, true);
    store_f16<8>(acc8, b0, 128, 0, lane);
    LDS_FENCE();

    // layer 4: [16,128]@[128,64] + bias -> scatter-add
#pragma unroll
    for (int i = 0; i < 4; ++i) acc4[i] = 0.0f;
    wmma_gemm<4, 4>(b0, w_ms_w2, acc4, lane);
    bias_act<4>(acc4, ms_b2, lane, false);
#pragma unroll
    for (int r = 0; r < 8; ++r) {
      int node = dst[e0 + r + 8 * half];
#pragma unroll
      for (int nt = 0; nt < 4; ++nt)
        atomicAdd(&aggr[(size_t)node * 64 + nt * 16 + n], acc4[nt][r]);
    }
    LDS_FENCE();
  }
}

// ---------------------------------------------------------------------------
// Node kernel (fused encoder + update), persistent waves, all 6 weight
// matrices (192KB f16) async-staged to LDS (CDNA5: 320KB/WGP makes this fit).
// LDS: weights 196608B, then 8 waves x 10240B act buffers = 278528B dynamic.
// ---------------------------------------------------------------------------
#define NW 8
__global__ void __launch_bounds__(256)
node_kernel(const float* __restrict__ x,
            const float* __restrict__ enc_b1, const float* __restrict__ enc_b2,
            const float* __restrict__ n1g, const float* __restrict__ n1b,
            const float* __restrict__ encs_b1, const float* __restrict__ encs_b2,
            const float* __restrict__ up_b1, const float* __restrict__ up_b2,
            const _Float16* __restrict__ zW,   // ws base: 98304 contiguous f16
            const float* __restrict__ aggr, const float* __restrict__ deg,
            float* __restrict__ out, int nNodes)
{
  extern __shared__ __align__(16) char smem[];
  _Float16* lds = (_Float16*)smem;
  const int tid = threadIdx.x;

  // Stage all node-path weights: 196608 B = 12288 x 16B chunks.
  for (int c = tid; c < 12288; c += 256)
    async_copy16((const char*)zW + (size_t)c * 16, smem + (size_t)c * 16);
  async_join();
  __syncthreads();

  const _Float16* w_enc_w1  = lds;
  const _Float16* w_enc_w2  = lds + 8192;
  const _Float16* w_encs_w1 = lds + 24576;
  const _Float16* w_encs_w2 = lds + 40960;
  const _Float16* w_up_w1   = lds + 57344;
  const _Float16* w_up_w2   = lds + 81920;

  const int wave = tid >> 5, lane = tid & 31;
  _Float16* A = lds + 98304 + wave * 5120;   // 16x128 f16
  _Float16* C = A + 2048;                    // 16x192 f16 (also hosts x tile)
  const int half = (lane >> 4) & 1, n = lane & 15;
  const int nTiles = nNodes >> 4;

  for (int tile = blockIdx.x * NW + wave; tile < nTiles; tile += gridDim.x * NW) {
    const int n0 = tile * 16;

    // x tile -> C (16x64, stride 64); dead before C is reused for concat
    for (int t = lane; t < 16 * 64; t += 32) {
      int m = t >> 6, c = t & 63;
      C[t] = (_Float16)x[(size_t)(n0 + m) * 64 + c];
    }
    LDS_FENCE();

    v8f acc[8];

    // L1: [16,64]@[64,128] + b, relu -> A
#pragma unroll
    for (int i = 0; i < 8; ++i) acc[i] = 0.0f;
    wmma_gemm<2, 8>(C, w_enc_w1, acc, lane);
    bias_act<8>(acc, enc_b1, lane, true);
    store_f16<8>(acc, A, 128, 0, lane);
    LDS_FENCE();

    // L2: [16,128]@[128,128] + b, LayerNorm -> C (stride 128)
#pragma unroll
    for (int i = 0; i < 8; ++i) acc[i] = 0.0f;
    wmma_gemm<4, 8>(A, w_enc_w2, acc, lane);
    ln_inplace<8>(acc, enc_b2, n1g, n1b, lane);
    store_f16<8>(acc, C, 128, 0, lane);
    LDS_FENCE();

    // L3: [16,128]@[128,128] + b, relu -> A
#pragma unroll
    for (int i = 0; i < 8; ++i) acc[i] = 0.0f;
    wmma_gemm<4, 8>(C, w_encs_w1, acc, lane);
    bias_act<8>(acc, encs_b1, lane, true);
    store_f16<8>(acc, A, 128, 0, lane);
    LDS_FENCE();

    // L4: [16,128]@[128,128] + b -> C cols 0..127 (stride 192)
#pragma unroll
    for (int i = 0; i < 8; ++i) acc[i] = 0.0f;
    wmma_gemm<4, 8>(A, w_encs_w2, acc, lane);
    bias_act<8>(acc, encs_b2, lane, false);
    store_f16<8>(acc, C, 192, 0, lane);

    // aggr tile -> C cols 128..191
    for (int t = lane; t < 16 * 64; t += 32) {
      int m = t >> 6, c = t & 63;
      C[m * 192 + 128 + c] = (_Float16)aggr[(size_t)(n0 + m) * 64 + c];
    }
    LDS_FENCE();

    // L5: [16,192]@[192,128] + b, relu -> A
#pragma unroll
    for (int i = 0; i < 8; ++i) acc[i] = 0.0f;
    wmma_gemm<6, 8>(C, w_up_w1, acc, lane);
    bias_act<8>(acc, up_b1, lane, true);
    store_f16<8>(acc, A, 128, 0, lane);
    LDS_FENCE();

    // L6: [16,128]@[128,128] + b, degree mask -> out
#pragma unroll
    for (int i = 0; i < 8; ++i) acc[i] = 0.0f;
    wmma_gemm<4, 8>(A, w_up_w2, acc, lane);
    bias_act<8>(acc, up_b2, lane, false);
#pragma unroll
    for (int r = 0; r < 8; ++r) {
      int node = n0 + r + 8 * half;
      float dv = deg[node];
#pragma unroll
      for (int nt = 0; nt < 8; ++nt) {
        float v = acc[nt][r];
        out[(size_t)node * 128 + nt * 16 + n] = (dv == 0.f) ? 0.f : v;
      }
    }
    LDS_FENCE();
  }
}

// ---------------------------------------------------------------------------
extern "C" void kernel_launch(void* const* d_in, const int* in_sizes, int n_in,
                              void* d_out, int out_size, void* d_ws, size_t ws_size,
                              hipStream_t stream) {
  const float* x         = (const float*)d_in[0];
  const int*   ei        = (const int*)d_in[1];
  const float* edge_attr = (const float*)d_in[2];
  const float* enc_w1  = (const float*)d_in[3];
  const float* enc_b1  = (const float*)d_in[4];
  const float* enc_w2  = (const float*)d_in[5];
  const float* enc_b2  = (const float*)d_in[6];
  const float* n1g     = (const float*)d_in[7];
  const float* n1b     = (const float*)d_in[8];
  const float* encs_w1 = (const float*)d_in[9];
  const float* encs_b1 = (const float*)d_in[10];
  const float* encs_w2 = (const float*)d_in[11];
  const float* encs_b2 = (const float*)d_in[12];
  const float* mm_w1   = (const float*)d_in[13];
  const float* mm_b1   = (const float*)d_in[14];
  const float* mm_w2   = (const float*)d_in[15];
  const float* mm_b2   = (const float*)d_in[16];
  const float* n2g     = (const float*)d_in[17];
  const float* n2b     = (const float*)d_in[18];
  const float* ms_w1   = (const float*)d_in[19];
  const float* ms_b1   = (const float*)d_in[20];
  const float* ms_w2   = (const float*)d_in[21];
  const float* ms_b2   = (const float*)d_in[22];
  const float* up_w1   = (const float*)d_in[23];
  const float* up_b1   = (const float*)d_in[24];
  const float* up_w2   = (const float*)d_in[25];
  const float* up_b2   = (const float*)d_in[26];

  const int nNodes = in_sizes[0] / 64;
  const int nEdges = in_sizes[2] / 3;

  _Float16* wsf16 = (_Float16*)d_ws;
  float* aggr = (float*)((char*)d_ws + AGGR_BYTE_OFF);
  float* deg  = aggr + (size_t)nNodes * 64;

  // 1) zero aggr + deg
  {
    long long n = (long long)nNodes * 64 + nNodes;
    int blocks = (int)((n + 255) / 256);
    if (blocks > 4096) blocks = 4096;
    zero_kernel<<<blocks, 256, 0, stream>>>(aggr, n);
  }

  // 2) swizzle weights into f16 B-fragment layout
  auto swz = [&](const float* src, size_t off, int K, int N) {
    int KT = K / 32, NT = N / 16;
    int threads = KT * NT * 32;
    swizzle_kernel<<<(threads + 127) / 128, 128, 0, stream>>>(src, wsf16 + off, KT, NT, N);
  };
  swz(mm_w2,   OFF_MM_W2,   128, 64);
  swz(ms_w1,   OFF_MS_W1,   64, 128);
  swz(ms_w2,   OFF_MS_W2,   128, 64);
  swz(enc_w1,  OFF_ENC_W1,  64, 128);
  swz(enc_w2,  OFF_ENC_W2,  128, 128);
  swz(encs_w1, OFF_ENCS_W1, 128, 128);
  swz(encs_w2, OFF_ENCS_W2, 128, 128);
  swz(up_w1,   OFF_UP_W1,   192, 128);
  swz(up_w2,   OFF_UP_W2,   128, 128);

  // 3) edge messages + scatter + degree (persistent, weights in LDS)
  edge_kernel<<<512, 128, 73728, stream>>>(
      edge_attr, ei, mm_w1, mm_b1, mm_b2, n2g, n2b, ms_b1, ms_b2,
      wsf16 + OFF_MM_W2, aggr, deg, nEdges);

  // 4) fused node encoder + update (persistent, weights in LDS)
  node_kernel<<<128, 256, 278528, stream>>>(
      x, enc_b1, enc_b2, n1g, n1b, encs_b1, encs_b2, up_b1, up_b2,
      wsf16 + OFF_ENC_W1, aggr, deg, (float*)d_out, nNodes);
}
